// NNUtil_29368986370119
// MI455X (gfx1250) — matrix operand order repaired and checked by reading, not compiled, not run on hardware
//
#include <hip/hip_runtime.h>
#include <hip/hip_bf16.h>

#define NPTS 8192
#define DIMS 512
#define KSEL 64
#define MQ   (NPTS - KSEL) /* 8128 */

typedef unsigned short u16;
typedef __attribute__((ext_vector_type(8)))  unsigned short ushort8;
typedef __attribute__((ext_vector_type(16))) __bf16         v16bf;
typedef __attribute__((ext_vector_type(8)))  float          v8f;

union FragBF { v16bf v; ushort8 h[2]; };

__device__ __forceinline__ u16 f2bf(float f) {
  unsigned u = __float_as_uint(f);
  unsigned r = u + 0x7fffu + ((u >> 16) & 1u);   // round-to-nearest-even
  return (u16)(r >> 16);
}
__device__ __forceinline__ float bf2f(u16 h) {
  return __uint_as_float(((unsigned)h) << 16);
}
// monotonic float <-> uint32 map (for max/min reductions on keys)
__device__ __forceinline__ unsigned ford(float f) {
  unsigned u = __float_as_uint(f);
  return (u & 0x80000000u) ? ~u : (u | 0x80000000u);
}
__device__ __forceinline__ float finv(unsigned o) {
  unsigned u = (o & 0x80000000u) ? (o & 0x7fffffffu) : ~o;
  return __uint_as_float(u);
}
__device__ __forceinline__ unsigned ballot32(bool p) {
#if __has_builtin(__builtin_amdgcn_ballot_w32)
  return __builtin_amdgcn_ballot_w32(p);
#else
  return (unsigned)__ballot(p);
#endif
}
__device__ __forceinline__ unsigned long long sx64(unsigned long long v, int m) {
  unsigned lo = (unsigned)__shfl_xor((int)(unsigned)v, m);
  unsigned hi = (unsigned)__shfl_xor((int)(unsigned)(v >> 32), m);
  return ((unsigned long long)hi << 32) | lo;
}
__device__ __forceinline__ unsigned long long wave_kmax(unsigned long long k) {
#pragma unroll
  for (int off = 16; off; off >>= 1) { unsigned long long o = sx64(k, off); if (o > k) k = o; }
  return k;
}
__device__ __forceinline__ unsigned long long wave_kmin(unsigned long long k) {
#pragma unroll
  for (int off = 16; off; off >>= 1) { unsigned long long o = sx64(k, off); if (o < k) k = o; }
  return k;
}
__device__ __forceinline__ unsigned wave_umax(unsigned v) {
#pragma unroll
  for (int off = 16; off; off >>= 1) { unsigned o = (unsigned)__shfl_xor((int)v, off); if (o > v) v = o; }
  return v;
}

// replace current wave-wide max slot (2 slots/lane) with (vs, js); refresh thr
__device__ __forceinline__ void topk_insert(float vs, int js, float (&d)[2], int (&ix)[2],
                                            float &thr, int lane) {
  int   msl = (d[1] > d[0]) ? 1 : 0;
  float mv  = d[msl];
  unsigned long long k = ((unsigned long long)ford(mv) << 6) |
                         (unsigned)(lane << 1) | (unsigned)msl;
  k = wave_kmax(k);
  float curmax = finv((unsigned)(k >> 6));
  if (vs < curmax) {
    if (lane == (int)((k >> 1) & 31u)) {
      int os = (int)(k & 1u);
      d[os] = vs; ix[os] = js;
    }
    unsigned t0 = ford(d[0]), t1 = ford(d[1]);
    thr = finv(wave_umax(t1 > t0 ? t1 : t0));
  }
}

// ---------------- prep: fp32 -> bf16 hi/lo planes + squared norms ----------------
__global__ void __launch_bounds__(32)
prep_kernel(const float* __restrict__ x, u16* __restrict__ xh, u16* __restrict__ xl,
            float* __restrict__ sq) {
  const int row = blockIdx.x;
  const int l   = threadIdx.x;
  const float* xr = x + (size_t)row * DIMS;
  float s = 0.0f;
  for (int e = l; e < DIMS; e += 32) {
    float f = xr[e];
    s = fmaf(f, f, s);
    u16 h = f2bf(f);
    xh[(size_t)row * DIMS + e] = h;
    xl[(size_t)row * DIMS + e] = f2bf(f - bf2f(h));
  }
#pragma unroll
  for (int off = 16; off; off >>= 1) s += __shfl_xor(s, off);
  if (l == 0) sq[row] = s;
}

// ---------------- main: causal GEMM (bf16x3 WMMA) + streaming top-64 ----------------
__global__ void __launch_bounds__(128)
knn_kernel(const u16* __restrict__ xh, const u16* __restrict__ xl,
           const float* __restrict__ sqg, const float* __restrict__ x,
           float* __restrict__ outd, long long* __restrict__ outi) {
  __shared__ u16   qh_s[16][DIMS];
  __shared__ u16   ql_s[16][DIMS];
  __shared__ float sqi_s[16];
  __shared__ float dtile[16][64];

  const float INFF = __int_as_float(0x7f800000);
  // heavy blocks (large i0, more j-tiles) first
  const int i0   = KSEL + 16 * (gridDim.x - 1 - blockIdx.x);
  const int w    = threadIdx.x >> 5;
  const int lane = threadIdx.x & 31;
  const int nloc = lane & 15;
  const int kh   = lane >> 4;

  // stage + split-convert the 16x512 query tile
  for (int e = threadIdx.x; e < 16 * DIMS; e += 128) {
    int m = e >> 9, d = e & (DIMS - 1);
    float f = x[(size_t)(i0 + m) * DIMS + d];
    u16 h = f2bf(f);
    qh_s[m][d] = h;
    ql_s[m][d] = f2bf(f - bf2f(h));
  }
  if (threadIdx.x < 16) sqi_s[threadIdx.x] = sqg[i0 + threadIdx.x];
  __syncthreads();

  // register-resident top-64 per row (4 rows/wave, 2 slots/lane)
  float td[4][2]; int ti[4][2]; float thr[4];
#pragma unroll
  for (int r = 0; r < 4; ++r) {
    td[r][0] = INFF; td[r][1] = INFF;
    ti[r][0] = 0;    ti[r][1] = 0;
    thr[r]   = INFF;
  }

  for (int jt = 0; jt <= i0 + 14; jt += 64) {
    const int j = jt + (w << 4) + nloc;            // this lane's B column / candidate row
    const u16* bhp = xh + (size_t)j * DIMS;
    const u16* blp = xl + (size_t)j * DIMS;
    if (j + 64 < NPTS) __builtin_prefetch(bhp + (size_t)64 * DIMS, 0, 0);

    v8f cH = {0.f,0.f,0.f,0.f,0.f,0.f,0.f,0.f};
    v8f cL = {0.f,0.f,0.f,0.f,0.f,0.f,0.f,0.f};
#pragma unroll 4
    for (int kk = 0; kk < DIMS; kk += 32) {
      FragBF ah, al, bh, bl;
      const int a0 = kk + kh * 8;                  // A: 16x32 bf16 per-lane layout
      ah.h[0] = *(const ushort8*)&qh_s[nloc][a0];
      ah.h[1] = *(const ushort8*)&qh_s[nloc][a0 + 16];
      al.h[0] = *(const ushort8*)&ql_s[nloc][a0];
      al.h[1] = *(const ushort8*)&ql_s[nloc][a0 + 16];
      const int b0 = kk + kh * 8;                  // B: 32x16 bf16 per-lane layout
      bh.h[0] = *(const ushort8*)&bhp[b0];
      bh.h[1] = *(const ushort8*)&bhp[b0 + 16];
      bl.h[0] = *(const ushort8*)&blp[b0];
      bl.h[1] = *(const ushort8*)&blp[b0 + 16];
      cH = __builtin_amdgcn_wmma_f32_16x16x32_bf16(false, ah.v, false, bh.v, (short)0, cH, false, false);
      cL = __builtin_amdgcn_wmma_f32_16x16x32_bf16(false, ah.v, false, bl.v, (short)0, cL, false, false);
      cL = __builtin_amdgcn_wmma_f32_16x16x32_bf16(false, al.v, false, bh.v, (short)0, cL, false, false);
    }

    // epilogue: d2 = |q|^2 + |x|^2 - 2 q.x, causal mask, stage to LDS
    const float sqj = sqg[j];
#pragma unroll
    for (int r = 0; r < 8; ++r) {
      int m = r + (kh << 3);
      float d2 = sqi_s[m] + sqj - 2.0f * (cH[r] + cL[r]);
      if (j >= i0 + m) d2 = INFF;
      dtile[m][(w << 4) + nloc] = d2;
    }
    __syncthreads();

    // streaming top-64 update: wave w owns rows 4w..4w+3
#pragma unroll
    for (int r = 0; r < 4; ++r) {
      const int row = (w << 2) + r;
#pragma unroll
      for (int p = 0; p < 2; ++p) {
        const int col = (p << 5) + lane;
        const float v = dtile[row][col];
        const int   jc = jt + col;
        unsigned mask = ballot32(v < thr[r]);
        while (mask) {
          int src = (int)__builtin_ctz(mask);
          float vs = __shfl(v, src);
          int   js = __shfl(jc, src);
          topk_insert(vs, js, td[r], ti[r], thr[r], lane);
          mask &= ~(1u << src);
          mask &= ballot32(v < thr[r]);
        }
      }
    }
    __syncthreads();
  }

  // selection-sort the 64 survivors ascending, write dist (f32) + idx (i64)
#pragma unroll
  for (int r = 0; r < 4; ++r) {
    const int gi = i0 + (w << 2) + r;
    const size_t ob = (size_t)(gi - KSEL) * KSEL;
    float a0 = td[r][0], a1 = td[r][1];
    int   j0 = ti[r][0], j1 = ti[r][1];
    for (int t = 0; t < KSEL; ++t) {
      unsigned long long k0 = ((unsigned long long)ford(a0) << 6) | (unsigned)(lane << 1);
      unsigned long long k1 = ((unsigned long long)ford(a1) << 6) | (unsigned)(lane << 1) | 1u;
      unsigned long long k  = (k1 < k0) ? k1 : k0;
      k = wave_kmin(k);
      const int ol = (int)((k >> 1) & 31u);
      const int os = (int)(k & 1u);
      const float dv = finv((unsigned)(k >> 6));
      const int   jv = __shfl(os ? j1 : j0, ol);
      if (lane == 0) { outd[ob + t] = dv; outi[ob + t] = (long long)jv; }
      if (lane == ol) { if (os) a1 = INFF; else a0 = INFF; }
    }
  }
}

extern "C" void kernel_launch(void* const* d_in, const int* in_sizes, int n_in,
                              void* d_out, int out_size, void* d_ws, size_t ws_size,
                              hipStream_t stream) {
  (void)in_sizes; (void)n_in; (void)out_size;
  const float* x = (const float*)d_in[0];   // d_in[1] is k==64 (compile-time constant here)

  // workspace layout: xh (bf16 hi), xl (bf16 lo), sq (f32 norms) ~16.8 MB
  u16*   xh  = (u16*)d_ws;
  u16*   xl  = xh + (size_t)NPTS * DIMS;
  float* sqg = (float*)(xl + (size_t)NPTS * DIMS);
  const size_t need = (size_t)NPTS * DIMS * 2 * sizeof(u16) + (size_t)NPTS * sizeof(float);
  if (ws_size < need) return;

  float*     outd = (float*)d_out;
  long long* outi = (long long*)((char*)d_out + (size_t)MQ * KSEL * sizeof(float));

  prep_kernel<<<NPTS, 32, 0, stream>>>(x, xh, xl, sqg);
  knn_kernel<<<MQ / 16, 128, 0, stream>>>(xh, xl, sqg, x, outd, outi);
}
